// AttentionHead_79010218377580
// MI455X (gfx1250) — matrix operand order repaired and checked by reading, not compiled, not run on hardware
//
#include <hip/hip_runtime.h>
#include <hip/hip_bf16.h>

// ---------------------------------------------------------------------------
// Single-head causal attention for MI455X (gfx1250), bf16 WMMA + fp32 accum.
// B=8, T=2048, D=1024, HS=64.
// ---------------------------------------------------------------------------

#define B_ 8
#define T_ 2048
#define D_ 1024
#define HS_ 64
#define BT_ (B_ * T_)

typedef __attribute__((ext_vector_type(16))) __bf16 v16bf;
typedef __attribute__((ext_vector_type(8)))  float  v8f;

union BFrag {
  unsigned u[8];
  uint4    q[2];
  v16bf    v;
};

// fp32 -> packed bf16x2; prefer the HW packed-convert if the toolchain has it.
#if __has_builtin(__builtin_amdgcn_cvt_pk_bf16_f32)
typedef __attribute__((ext_vector_type(2))) __bf16 v2bf;
static __device__ __forceinline__ unsigned pkbf(float lo, float hi) {
  union { v2bf v; unsigned u; } c;
  c.v = __builtin_amdgcn_cvt_pk_bf16_f32(lo, hi);
  return c.u;
}
#else
static __device__ __forceinline__ unsigned pkbf(float lo, float hi) {
  unsigned ul = __float_as_uint(lo);
  ul += 0x7fffu + ((ul >> 16) & 1u);   // round to nearest even
  unsigned uh = __float_as_uint(hi);
  uh += 0x7fffu + ((uh >> 16) & 1u);
  return (ul >> 16) | (uh & 0xffff0000u);
}
#endif

static __device__ __forceinline__ unsigned short f2bf(float f) {
  unsigned u = __float_as_uint(f);
  u += 0x7fffu + ((u >> 16) & 1u);
  return (unsigned short)(u >> 16);
}

// Async copy 16 bytes global -> LDS (gfx1250, tracked by ASYNCcnt).
static __device__ __forceinline__ void async_load_b128(unsigned lds_off,
                                                       const void* gptr) {
  asm volatile("global_load_async_to_lds_b128 %0, %1, off"
               :
               : "v"(lds_off), "v"((unsigned long long)(uintptr_t)gptr)
               : "memory");
}

// ---------------------------------------------------------------------------
// Kernel 1: convert+transpose weights: Wt[w][c][k] = W_w[k][c] in bf16.
// ---------------------------------------------------------------------------
__global__ void prep_w_kernel(const float* __restrict__ Wq,
                              const float* __restrict__ Wk,
                              const float* __restrict__ Wv,
                              unsigned short* __restrict__ Wt) {
  int idx = blockIdx.x * 256 + threadIdx.x;
  if (idx >= 3 * HS_ * D_) return;
  int k = idx & (D_ - 1);
  int c = (idx >> 10) & (HS_ - 1);
  int w = idx >> 16;
  const float* W = (w == 0) ? Wq : (w == 1) ? Wk : Wv;
  Wt[idx] = f2bf(W[k * HS_ + c]);
}

// ---------------------------------------------------------------------------
// Kernel 2: QKV projection. One wave per 16-row tile of x.
//   Q,K -> bf16 row-major [BT][64] (via LDS transpose for coalesced stores)
//   V   -> bf16 transposed [B][64][T] (per-lane column pack, direct b128)
// ---------------------------------------------------------------------------
__global__ __launch_bounds__(128) void qkv_proj_kernel(
    const float* __restrict__ x, const unsigned short* __restrict__ Wt,
    const float* __restrict__ bq, const float* __restrict__ bk,
    const float* __restrict__ bv,
    unsigned short* __restrict__ Qb, unsigned short* __restrict__ Kb,
    unsigned short* __restrict__ Vt) {
  __shared__ __align__(16) unsigned short stage[4][16][64];

  const int lane = threadIdx.x & 31;
  const int wv   = threadIdx.x >> 5;
  const int tile = blockIdx.x * 4 + wv;
  const int t0   = tile * 16;          // global row in [0, 16384)
  const int mrow = lane & 15;
  const int hi   = lane >> 4;          // lane half (A/B fragment K split)

  v8f acc[3][4];
  const v8f zf = {0.f, 0.f, 0.f, 0.f, 0.f, 0.f, 0.f, 0.f};
#pragma unroll
  for (int w = 0; w < 3; ++w)
#pragma unroll
    for (int nt = 0; nt < 4; ++nt) acc[w][nt] = zf;

  const float* xrow = x + (size_t)(t0 + mrow) * D_;

  for (int k0 = 0; k0 < D_; k0 += 32) {
    // A fragment: lane = row, K chunks [lkb..lkb+7] and [lkb+16..lkb+23]
    BFrag a;
    const float* p = xrow + k0 + hi * 8;
    float4 c0 = *(const float4*)(p + 0);
    float4 c1 = *(const float4*)(p + 4);
    float4 c2 = *(const float4*)(p + 16);
    float4 c3 = *(const float4*)(p + 20);
    a.u[0] = pkbf(c0.x, c0.y); a.u[1] = pkbf(c0.z, c0.w);
    a.u[2] = pkbf(c1.x, c1.y); a.u[3] = pkbf(c1.z, c1.w);
    a.u[4] = pkbf(c2.x, c2.y); a.u[5] = pkbf(c2.z, c2.w);
    a.u[6] = pkbf(c3.x, c3.y); a.u[7] = pkbf(c3.z, c3.w);

#pragma unroll
    for (int nt = 0; nt < 4; ++nt) {
      // B fragments for Q/K/V weights.
      BFrag bm[3];
      const unsigned short* wp =
          Wt + ((size_t)(nt * 16 + mrow)) * D_ + k0 + hi * 8;
#pragma unroll
      for (int w = 0; w < 3; ++w) {
        bm[w].q[0] = *(const uint4*)(wp + w * (HS_ * D_));
        bm[w].q[1] = *(const uint4*)(wp + w * (HS_ * D_) + 16);
      }
#pragma unroll
      for (int w = 0; w < 3; ++w)
        acc[w][nt] = __builtin_amdgcn_wmma_f32_16x16x32_bf16(
            false, a.v, false, bm[w].v, (short)0, acc[w][nt], false, false);
    }
  }

  // bias add (C/D layout: lane holds one column across 8 row-VGPRs)
#pragma unroll
  for (int nt = 0; nt < 4; ++nt) {
    const int c = nt * 16 + mrow;
    float b0 = bq[c], b1 = bk[c], b2 = bv[c];
#pragma unroll
    for (int r = 0; r < 8; ++r) {
      acc[0][nt][r] += b0;
      acc[1][nt][r] += b1;
      acc[2][nt][r] += b2;
    }
  }

  // V store (transposed): lane's 8 row values are consecutive along T
  {
    const int b  = t0 >> 11;
    const int tl = t0 & (T_ - 1);
#pragma unroll
    for (int nt = 0; nt < 4; ++nt) {
      uint4 pv;
      pv.x = pkbf(acc[2][nt][0], acc[2][nt][1]);
      pv.y = pkbf(acc[2][nt][2], acc[2][nt][3]);
      pv.z = pkbf(acc[2][nt][4], acc[2][nt][5]);
      pv.w = pkbf(acc[2][nt][6], acc[2][nt][7]);
      *(uint4*)(Vt + ((size_t)(b * HS_ + nt * 16 + mrow)) * T_ + tl + hi * 8) = pv;
    }
  }

  // Q,K stores: scatter C-layout into LDS row-major, read back coalesced
#pragma unroll
  for (int w = 0; w < 2; ++w) {
#pragma unroll
    for (int nt = 0; nt < 4; ++nt)
#pragma unroll
      for (int r = 0; r < 8; ++r)
        stage[wv][r + 8 * hi][nt * 16 + mrow] = f2bf(acc[w][nt][r]);

    asm volatile("s_wait_dscnt 0" ::: "memory");

    unsigned short* dst = (w == 0 ? Qb : Kb) + (size_t)t0 * HS_;
#pragma unroll
    for (int i = 0; i < 4; ++i) {
      int idx = i * 32 + lane;          // 16-byte chunk id (128 per tile)
      int row = idx >> 3, c8 = idx & 7;
      uint4 val = *(const uint4*)&stage[wv][row][c8 * 8];
      *(uint4*)(dst + row * HS_ + c8 * 8) = val;
    }
    asm volatile("s_wait_dscnt 0" ::: "memory");
  }
}

// ---------------------------------------------------------------------------
// Kernel 3: causal flash attention. 4 waves (adjacent 16-query tiles) per
// workgroup cooperatively stage each 32-key K/V block into LDS with
// double-buffered async copies; each wave runs online softmax + WMMA.
// ---------------------------------------------------------------------------
__global__ __launch_bounds__(128) void flash_attn_kernel(
    const unsigned short* __restrict__ Qb, const unsigned short* __restrict__ Kb,
    const unsigned short* __restrict__ Vt, float* __restrict__ out) {
  __shared__ __align__(16) unsigned short pl[4][16][32];      // P bounce (per wave)
  __shared__ __align__(16) unsigned short Kblk[2][32][64];    // K block, row-major
  __shared__ __align__(16) unsigned short Vblk[2][64][32];    // V^T block [hs][key]

  const int tid  = threadIdx.x;
  const int lane = tid & 31;
  const int wv   = tid >> 5;
  const int g    = blockIdx.x * 4 + wv;
  const int t0g  = g * 16;             // global query row
  const int b    = t0g >> 11;
  const int t0   = t0g & (T_ - 1);     // query row within batch
  const int mrow = lane & 15;
  const int hi   = lane >> 4;

  const int t0first = (blockIdx.x * 64) & (T_ - 1);
  const int nmax = (t0first + 95) >> 5;   // uniform block-loop trip count
  const int nblk = (t0 + 47) >> 5;        // this wave's causal bound

  // Q A-fragments for HS chunks [0..31], [32..63]
  BFrag qa[2];
#pragma unroll
  for (int h = 0; h < 2; ++h) {
    const unsigned short* qp = Qb + (size_t)(t0g + mrow) * HS_ + h * 32 + hi * 8;
    qa[h].q[0] = *(const uint4*)qp;
    qa[h].q[1] = *(const uint4*)(qp + 16);
  }

  v8f o[4];
  const v8f zf = {0.f, 0.f, 0.f, 0.f, 0.f, 0.f, 0.f, 0.f};
#pragma unroll
  for (int nt = 0; nt < 4; ++nt) o[nt] = zf;
  float mr[8], lr[8];
#pragma unroll
  for (int r = 0; r < 8; ++r) { mr[r] = -__builtin_inff(); lr[r] = 0.f; }

  // Cooperative stage of one K/V block: 2x256 chunks of 16B, 4 per thread.
  auto stage_kv = [&](int kbase, int buf) {
#pragma unroll
    for (int i = 0; i < 2; ++i) {
      int c = i * 128 + tid;
      int key = c >> 3, c8 = c & 7;
      async_load_b128((unsigned)(uintptr_t)&Kblk[buf][key][c8 * 8],
                      Kb + ((size_t)(b * T_ + kbase + key)) * HS_ + c8 * 8);
      int hs = c >> 2, c4 = c & 3;
      async_load_b128((unsigned)(uintptr_t)&Vblk[buf][hs][c4 * 8],
                      Vt + ((size_t)(b * HS_ + hs)) * T_ + kbase + c4 * 8);
    }
  };

  stage_kv(0, 0);                       // prime the pipeline

  for (int jb = 0; jb < nmax; ++jb) {
    const int kbase = jb * 32;
    const int buf   = jb & 1;

    if (jb + 1 < nmax) {
      stage_kv(kbase + 32, buf ^ 1);    // overlap next block's copy
      asm volatile("s_wait_asynccnt 4" ::: "memory");  // current block done
    } else {
      asm volatile("s_wait_asynccnt 0" ::: "memory");
    }
    __syncthreads();                    // block visible to all 4 waves

    if (jb < nblk) {
      // S = Q @ K^T : two 16-key column tiles, K-dim 64 = 2 WMMAs each
      v8f s[2];
#pragma unroll
      for (int ct = 0; ct < 2; ++ct) {
        s[ct] = zf;
#pragma unroll
        for (int h = 0; h < 2; ++h) {
          BFrag kf;
          const unsigned short* kp = &Kblk[buf][ct * 16 + mrow][h * 32 + hi * 8];
          kf.q[0] = *(const uint4*)kp;
          kf.q[1] = *(const uint4*)(kp + 16);
          s[ct] = __builtin_amdgcn_wmma_f32_16x16x32_bf16(
              false, qa[h].v, false, kf.v, (short)0, s[ct], false, false);
        }
      }

      // mask (pre-scale equivalent), online softmax update
      float al[8];
#pragma unroll
      for (int r = 0; r < 8; ++r) {
        const int q = t0 + r + hi * 8;
#pragma unroll
        for (int ct = 0; ct < 2; ++ct) {
          const int key = kbase + ct * 16 + mrow;
          float v = s[ct][r] * 0.125f;            // 1/sqrt(64)
          s[ct][r] = (key <= q) ? v : -__builtin_inff();
        }
        float rm = fmaxf(s[0][r], s[1][r]);
        rm = fmaxf(rm, __shfl_xor(rm, 1, 16));
        rm = fmaxf(rm, __shfl_xor(rm, 2, 16));
        rm = fmaxf(rm, __shfl_xor(rm, 4, 16));
        rm = fmaxf(rm, __shfl_xor(rm, 8, 16));
        float mnew = fmaxf(mr[r], rm);
        al[r] = __expf(mr[r] - mnew);
        float p0 = __expf(s[0][r] - mnew);
        float p1 = __expf(s[1][r] - mnew);
        float ps = p0 + p1;
        ps += __shfl_xor(ps, 1, 16);
        ps += __shfl_xor(ps, 2, 16);
        ps += __shfl_xor(ps, 4, 16);
        ps += __shfl_xor(ps, 8, 16);
        lr[r] = lr[r] * al[r] + ps;
        mr[r] = mnew;
        // P tile row-major in LDS (C-layout scatter)
        pl[wv][r + 8 * hi][mrow]      = f2bf(p0);
        pl[wv][r + 8 * hi][16 + mrow] = f2bf(p1);
      }
      asm volatile("s_wait_dscnt 0" ::: "memory");

      // rebuild P as A-fragment (16x32)
      BFrag pa;
      {
        const unsigned short* pp = &pl[wv][mrow][hi * 8];
        pa.q[0] = *(const uint4*)pp;
        pa.q[1] = *(const uint4*)(pp + 16);
      }

      // O = alpha*O + P @ V  (V^T in LDS gives contiguous K per lane)
#pragma unroll
      for (int nt = 0; nt < 4; ++nt) {
#pragma unroll
        for (int r = 0; r < 8; ++r) o[nt][r] *= al[r];
        BFrag vf;
        const unsigned short* vp = &Vblk[buf][nt * 16 + mrow][hi * 8];
        vf.q[0] = *(const uint4*)vp;
        vf.q[1] = *(const uint4*)(vp + 16);
        o[nt] = __builtin_amdgcn_wmma_f32_16x16x32_bf16(
            false, pa.v, false, vf.v, (short)0, o[nt], false, false);
      }
    }
    __syncthreads();                    // all waves done reading this buffer
  }

  // epilogue: normalize and store fp32 output
#pragma unroll
  for (int nt = 0; nt < 4; ++nt)
#pragma unroll
    for (int r = 0; r < 8; ++r) {
      const int m = r + 8 * hi;
      out[(size_t)(t0g + m) * HS_ + nt * 16 + mrow] = o[nt][r] / lr[r];
    }
}

// ---------------------------------------------------------------------------
extern "C" void kernel_launch(void* const* d_in, const int* in_sizes, int n_in,
                              void* d_out, int out_size, void* d_ws, size_t ws_size,
                              hipStream_t stream) {
  const float* x  = (const float*)d_in[0];
  const float* Wq = (const float*)d_in[1];
  const float* bq = (const float*)d_in[2];
  const float* Wk = (const float*)d_in[3];
  const float* bk = (const float*)d_in[4];
  const float* Wv = (const float*)d_in[5];
  const float* bv = (const float*)d_in[6];
  float* out = (float*)d_out;

  char* ws = (char*)d_ws;
  unsigned short* Wt = (unsigned short*)(ws);                         // 384 KB
  unsigned short* Qb = (unsigned short*)(ws + (512 << 10));           // 2 MB
  unsigned short* Kb = (unsigned short*)(ws + (512 << 10) + (2 << 20));
  unsigned short* Vt = (unsigned short*)(ws + (512 << 10) + (4 << 20));

  prep_w_kernel<<<dim3((3 * HS_ * D_ + 255) / 256), dim3(256), 0, stream>>>(
      Wq, Wk, Wv, Wt);
  qkv_proj_kernel<<<dim3(BT_ / 16 / 4), dim3(128), 0, stream>>>(
      x, Wt, bq, bk, bv, Qb, Kb, Vt);
  flash_attn_kernel<<<dim3(BT_ / 16 / 4), dim3(128), 0, stream>>>(
      Qb, Kb, Vt, out);
}
